// Model_80917183857084
// MI455X (gfx1250) — matrix operand order repaired and checked by reading, not compiled, not run on hardware
//
#include <hip/hip_runtime.h>
#include <hip/hip_bf16.h>
#include <cstring>
#include <math.h>

// ---------------------------------------------------------------------------
// Types for CDNA5 WMMA (wave32, 16x16x32 f16 -> f32)
// ---------------------------------------------------------------------------
typedef __attribute__((ext_vector_type(16))) _Float16 v16h;
typedef __attribute__((ext_vector_type(8)))  float    v8f;

#define BB   16
#define NN   2048
#define SS   256
#define DD   256
#define HH   4
#define DHH  64

// ---------------------------------------------------------------------------
// Generic batched WMMA GEMM:  C = act(bn(alpha*A@B^T (+C) (+bias) (+rowres))) (+res)
// A: [M,K] row-major (lda).  B: weights [N,K] (b_kn=0) or [K,N] (b_kn=1).
// Batched via z with outer/inner strides (z -> zo=z/zdiv, zi=z%zdiv).
// LDS tiles are stored pre-swizzled in fragment-ready f16 layout so the inner
// loop is pure ds_load_b128 -> v_wmma; staging has a branchless fast path for
// full interior tiles so global loads pipeline without per-element waits.
// ---------------------------------------------------------------------------
struct GemmParams {
  const float* A;  long lda;  long sAo, sAi;
  const float* Bw; long ldb;  long sBo, sBi;  int b_kn;
  float*       C;  long ldc;  long sCo, sCi;
  int M, N, K, zdiv;
  float alpha;
  int load_c;
  const float* bias;                                   // per-n, may be null
  const float* bng; const float* bnb;                  // batchnorm fold (per-n)
  const float* bnm; const float* bnv;                  // null => no BN
  int act;                                             // 0 none,1 relu,2 lrelu,3 gelu,4 sigmoid
  const float* res; long ldres; int resmode; long res_div; // 1: post-act res[m], 2: pre-BN res[m/div]
  int amax; long amax_div;                             // atomic max store: C[(m/div)*ldc+n]
};

__global__ __launch_bounds__(128) void gemm_wmma_kernel(GemmParams p) {
  // fragment-ready f16 tiles: aF[wave][lane][16 halves], bF[ctile][lane][16]
  __shared__ __align__(32) _Float16 aF[4 * 32 * 16];
  __shared__ __align__(32) _Float16 bF[4 * 32 * 16];

  int z  = blockIdx.z;
  int zo = z / p.zdiv, zi = z - zo * p.zdiv;
  const float* A  = p.A  + (long)zo * p.sAo + (long)zi * p.sAi;
  const float* Bm = p.Bw + (long)zo * p.sBo + (long)zi * p.sBi;
  float*       C  = p.C  + (long)zo * p.sCo + (long)zi * p.sCi;

  int m0   = blockIdx.y * 64;
  int n0   = blockIdx.x * 64;
  int tid  = threadIdx.x;
  int wid  = tid >> 5;
  int lane = tid & 31;
  int g    = lane >> 4;
  int l16  = lane & 15;

  v8f acc[4];
  #pragma unroll
  for (int c = 0; c < 4; ++c)
    #pragma unroll
    for (int r = 0; r < 8; ++r) acc[c][r] = 0.f;

  bool fullM = (m0 + 64 <= p.M);
  bool fullN = (n0 + 64 <= p.N);

  for (int k0 = 0; k0 < p.K; k0 += 32) {
    bool fullK = (k0 + 32 <= p.K);
    // ---- stage A tile (64x32) into swizzled f16 fragment layout ----
    // element (r, k): wave=r>>4, m=r&15; g2=(k>>3)&1, j=((k>>4)<<3)|(k&7)
    if (fullM && fullK) {
      #pragma unroll
      for (int t = 0; t < 16; ++t) {
        int i = tid + t * 128;
        int r = i >> 5, cc = i & 31;
        float v = A[(long)(m0 + r) * p.lda + (k0 + cc)];
        int wv = r >> 4, mm = r & 15;
        int g2 = (cc >> 3) & 1;
        int j  = ((cc >> 4) << 3) | (cc & 7);
        aF[((wv << 5) + (g2 << 4) + mm) * 16 + j] = (_Float16)v;
      }
    } else {
      for (int i = tid; i < 64 * 32; i += 128) {
        int r = i >> 5, cc = i & 31;
        int gm = m0 + r, gk = k0 + cc;
        float v = (gm < p.M && gk < p.K) ? A[(long)gm * p.lda + gk] : 0.f;
        int wv = r >> 4, mm = r & 15;
        int g2 = (cc >> 3) & 1;
        int j  = ((cc >> 4) << 3) | (cc & 7);
        aF[((wv << 5) + (g2 << 4) + mm) * 16 + j] = (_Float16)v;
      }
    }
    // ---- stage B tile (64n x 32k) into swizzled f16 fragment layout ----
    // element (n, k): ctile=n>>4, nl=n&15; g2=(k>>3)&1, j=((k>>4)<<3)|(k&7)
    if (fullN && fullK) {
      #pragma unroll
      for (int t = 0; t < 16; ++t) {
        int i = tid + t * 128;
        int r = i >> 5, cc = i & 31;
        float v = p.b_kn ? Bm[(long)(k0 + cc) * p.ldb + (n0 + r)]
                         : Bm[(long)(n0 + r) * p.ldb + (k0 + cc)];
        int ct = r >> 4, nl = r & 15;
        int g2 = (cc >> 3) & 1;
        int j  = ((cc >> 4) << 3) | (cc & 7);
        bF[((ct << 5) + (g2 << 4) + nl) * 16 + j] = (_Float16)v;
      }
    } else {
      for (int i = tid; i < 64 * 32; i += 128) {
        int r = i >> 5, cc = i & 31;
        int gn = n0 + r, gk = k0 + cc;
        float v = 0.f;
        if (gn < p.N && gk < p.K)
          v = p.b_kn ? Bm[(long)gk * p.ldb + gn] : Bm[(long)gn * p.ldb + gk];
        int ct = r >> 4, nl = r & 15;
        int g2 = (cc >> 3) & 1;
        int j  = ((cc >> 4) << 3) | (cc & 7);
        bF[((ct << 5) + (g2 << 4) + nl) * 16 + j] = (_Float16)v;
      }
    }
    // prefetch next K tile (speculative; dropped if invalid)
    if (k0 + 32 < p.K) {
      int r = tid >> 1;
      if (m0 + r < p.M) __builtin_prefetch(&A[(long)(m0 + r) * p.lda + k0 + 32], 0, 1);
    }
    __syncthreads();

    // inner loop: contiguous 32B fragment loads feeding WMMA directly
    v16h af = *(const v16h*)&aF[((wid << 5) + lane) * 16];
    #pragma unroll
    for (int c = 0; c < 4; ++c) {
      v16h bf = *(const v16h*)&bF[((c << 5) + lane) * 16];
      acc[c] = __builtin_amdgcn_wmma_f32_16x16x32_f16(
          false, af, false, bf, (short)0, acc[c], false, false);
    }
    __syncthreads();
  }

  // epilogue: C rows M = wid*16 + g*8 + r, col N = c*16 + l16
  bool has_bn = (p.bng != nullptr);
  #pragma unroll
  for (int c = 0; c < 4; ++c) {
    int gn = n0 + c * 16 + l16;
    if (gn >= p.N) continue;
    float bnscale = 1.f, bnshift = 0.f;
    if (has_bn) {
      bnscale = p.bng[gn] * rsqrtf(p.bnv[gn] + 1e-5f);
      bnshift = p.bnb[gn] - p.bnm[gn] * bnscale;
    }
    float bias = p.bias ? p.bias[gn] : 0.f;
    #pragma unroll
    for (int r = 0; r < 8; ++r) {
      int gm = m0 + wid * 16 + g * 8 + r;
      if (gm >= p.M) continue;
      long cidx = (long)gm * p.ldc + gn;
      float t = p.alpha * acc[c][r];
      if (p.load_c) t += C[cidx];
      t += bias;
      if (p.res && p.resmode == 2) t += p.res[(long)(gm / p.res_div) * p.ldres + gn];
      if (has_bn) t = t * bnscale + bnshift;
      switch (p.act) {
        case 1: t = fmaxf(t, 0.f); break;
        case 2: t = t > 0.f ? t : 0.2f * t; break;
        case 3: t = 0.5f * t * (1.f + erff(t * 0.70710678118f)); break;
        case 4: t = 1.f / (1.f + expf(-t)); break;
        default: break;
      }
      if (p.res && p.resmode == 1) t += p.res[(long)gm * p.ldres + gn];
      if (p.amax) {
        atomicMax((int*)(C + (long)(gm / p.amax_div) * p.ldc + gn), __float_as_int(t));
      } else {
        C[cidx] = t;
      }
    }
  }
}

// ---------------------------------------------------------------------------
// Elementwise / reduction helper kernels
// ---------------------------------------------------------------------------
__global__ void zero_kernel(float* p, long n) {
  long t = (long)blockIdx.x * 256 + threadIdx.x;
  if (t < n) p[t] = 0.f;
}

__global__ void transpose_x_kernel(const float* x, float* xt) {
  long t = (long)blockIdx.x * 256 + threadIdx.x;
  long total = (long)BB * 3 * NN;
  if (t >= total) return;
  int n = (int)(t % NN); long r = t / NN; int c = (int)(r % 3); int b = (int)(r / 3);
  xt[((long)b * NN + n) * 3 + c] = x[t];
}

__global__ void fps_kernel(const float* xyz, int* fps_idx) {
  int b = blockIdx.x;
  const float* X = xyz + (long)b * NN * 3;
  __shared__ float dist[NN];
  __shared__ float rmax[256];
  __shared__ int   rid[256];
  __shared__ float cx, cy, cz;
  __shared__ int   far_s;
  int tid = threadIdx.x;
  for (int i = tid; i < NN; i += 256) dist[i] = 1e10f;
  if (tid == 0) far_s = 0;
  __syncthreads();
  for (int it = 0; it < SS; ++it) {
    if (tid == 0) {
      int f = far_s;
      fps_idx[b * SS + it] = f;
      cx = X[f * 3]; cy = X[f * 3 + 1]; cz = X[f * 3 + 2];
    }
    __syncthreads();
    float bm = -1.f; int bi = 0;
    for (int i = tid; i < NN; i += 256) {
      float dx = X[i * 3] - cx, dy = X[i * 3 + 1] - cy, dz = X[i * 3 + 2] - cz;
      float d = dx * dx + dy * dy + dz * dz;
      float nd = fminf(dist[i], d);
      dist[i] = nd;
      if (nd > bm) { bm = nd; bi = i; }
    }
    rmax[tid] = bm; rid[tid] = bi;
    __syncthreads();
    for (int s = 128; s > 0; s >>= 1) {
      if (tid < s) {
        if (rmax[tid + s] > rmax[tid] ||
            (rmax[tid + s] == rmax[tid] && rid[tid + s] < rid[tid])) {
          rmax[tid] = rmax[tid + s]; rid[tid] = rid[tid + s];
        }
      }
      __syncthreads();
    }
    if (tid == 0) far_s = rid[0];
    __syncthreads();
  }
}

__global__ void gather_fps_kernel(const float* xyz, const float* feats, const int* fps_idx,
                                  float* new_xyz, float* np_all) {
  int t = blockIdx.x * 256 + threadIdx.x;   // B*S
  if (t >= BB * SS) return;
  int b = t >> 8;
  int src = fps_idx[t];
  const float* xp = xyz + ((long)b * NN + src) * 3;
  float* nx = new_xyz + (long)t * 3;
  nx[0] = xp[0]; nx[1] = xp[1]; nx[2] = xp[2];
  const float* fp = feats + ((long)b * NN + src) * 64;
  float* np = np_all + (long)t * 64;
  for (int c = 0; c < 64; ++c) np[c] = fp[c];
}

// One query per block: bitonic sort of 2048 distances, keep nearest 256.
__global__ void knn_kernel(const float* xyz, const float* new_xyz, int* knn_idx) {
  int s = blockIdx.x, b = blockIdx.y;
  __shared__ float d[NN];
  __shared__ int   id[NN];
  const float* X = xyz + (long)b * NN * 3;
  const float* q = new_xyz + ((long)b * SS + s) * 3;
  float qx = q[0], qy = q[1], qz = q[2];
  int tid = threadIdx.x;
  for (int i = tid; i < NN; i += 256) {
    float dx = X[i * 3] - qx, dy = X[i * 3 + 1] - qy, dz = X[i * 3 + 2] - qz;
    d[i] = dx * dx + dy * dy + dz * dz;
    id[i] = i;
  }
  __syncthreads();
  for (int k = 2; k <= NN; k <<= 1) {
    for (int j = k >> 1; j > 0; j >>= 1) {
      for (int t = tid; t < NN; t += 256) {
        int ixj = t ^ j;
        if (ixj > t) {
          bool up = ((t & k) == 0);
          bool sw = up ? (d[t] > d[ixj]) : (d[t] < d[ixj]);
          if (sw) {
            float td = d[t]; d[t] = d[ixj]; d[ixj] = td;
            int   ti = id[t]; id[t] = id[ixj]; id[ixj] = ti;
          }
        }
      }
      __syncthreads();
    }
  }
  if (tid < SS) knn_idx[(((long)b * SS) + s) * SS + tid] = id[tid];
}

__global__ void w1diff_kernel(const float* w1, float* wd) {
  int t = blockIdx.x * 256 + threadIdx.x;   // 256*64
  if (t >= 256 * 64) return;
  int c = t & 63, o = t >> 6;
  wd[t] = w1[o * 128 + 64 + c] - w1[o * 128 + c];
}

__global__ void gather_knn_kernel(const float* feats, const int* knn_idx, float* ga, int b0) {
  long t = (long)blockIdx.x * 256 + threadIdx.x;
  long total = 2L * SS * SS * 64;
  if (t >= total) return;
  int c = (int)(t & 63);
  long r = t >> 6;                 // (bb*256 + s)*256 + k
  int k = (int)(r & 255);
  long bs = r >> 8;                // bb*256 + s
  int bglob = b0 + (int)(bs >> 8);
  int s = (int)(bs & 255);
  int idx = knn_idx[(((long)bglob * SS) + s) * SS + k];
  ga[t] = feats[((long)bglob * NN + idx) * 64 + c];
}

__global__ void layernorm_kernel(const float* x, float* y, const float* g, const float* b) {
  int row = blockIdx.x;                       // D = 256
  const float* xr = x + (long)row * DD;
  float* yr = y + (long)row * DD;
  int tid = threadIdx.x;
  __shared__ float red[256];
  float v = xr[tid];
  red[tid] = v; __syncthreads();
  for (int s = 128; s > 0; s >>= 1) { if (tid < s) red[tid] += red[tid + s]; __syncthreads(); }
  float mu = red[0] * (1.f / DD);
  __syncthreads();
  float d = v - mu;
  red[tid] = d * d; __syncthreads();
  for (int s = 128; s > 0; s >>= 1) { if (tid < s) red[tid] += red[tid + s]; __syncthreads(); }
  float var = red[0] * (1.f / DD);
  yr[tid] = g[tid] * d * rsqrtf(var + 1e-5f) + b[tid];
}

__global__ void softmax256_kernel(float* x) {
  long row = blockIdx.x;
  float* xr = x + row * 256;
  int tid = threadIdx.x;
  __shared__ float red[256];
  float v = xr[tid];
  red[tid] = v; __syncthreads();
  for (int s = 128; s > 0; s >>= 1) { if (tid < s) red[tid] = fmaxf(red[tid], red[tid + s]); __syncthreads(); }
  float mx = red[0];
  __syncthreads();
  float e = expf(v - mx);
  red[tid] = e; __syncthreads();
  for (int s = 128; s > 0; s >>= 1) { if (tid < s) red[tid] += red[tid + s]; __syncthreads(); }
  xr[tid] = e / red[0];
}

__global__ void pool_kernel(const float* feat, float* fmx, float* favg) {
  int b = blockIdx.x;
  int c = blockIdx.y * 256 + threadIdx.x;     // 768
  if (c >= 768) return;
  float mx = -1e30f, sm = 0.f;
  for (int s = 0; s < SS; ++s) {
    float v = feat[((long)b * SS + s) * 768 + c];
    mx = fmaxf(mx, v); sm += v;
  }
  fmx[b * 768 + c] = mx;
  favg[b * 768 + c] = sm * (1.f / SS);
}

__global__ void three_nn_kernel(const float* xyz, const float* new_xyz, int* idx3, float* w3) {
  int b = blockIdx.x;
  int n = blockIdx.y * 256 + threadIdx.x;
  __shared__ float nx[SS * 3];
  for (int i = threadIdx.x; i < SS * 3; i += 256) nx[i] = new_xyz[(long)b * SS * 3 + i];
  __syncthreads();
  const float* pp = xyz + ((long)b * NN + n) * 3;
  float px = pp[0], py = pp[1], pz = pp[2];
  float d0 = 1e30f, d1 = 1e30f, d2 = 1e30f; int i0 = 0, i1 = 0, i2 = 0;
  for (int s = 0; s < SS; ++s) {
    float dx = nx[s * 3] - px, dy = nx[s * 3 + 1] - py, dz = nx[s * 3 + 2] - pz;
    float d = dx * dx + dy * dy + dz * dz;
    if (d < d0)      { d2 = d1; i2 = i1; d1 = d0; i1 = i0; d0 = d; i0 = s; }
    else if (d < d1) { d2 = d1; i2 = i1; d1 = d; i1 = s; }
    else if (d < d2) { d2 = d; i2 = s; }
  }
  float w0 = 1.f / (d0 + 1e-8f), w1 = 1.f / (d1 + 1e-8f), w2 = 1.f / (d2 + 1e-8f);
  float ws = w0 + w1 + w2;
  long o = ((long)b * NN + n) * 3;
  idx3[o] = i0; idx3[o + 1] = i1; idx3[o + 2] = i2;
  w3[o] = w0 / ws; w3[o + 1] = w1 / ws; w3[o + 2] = w2 / ws;
}

__global__ void interp_kernel(const float* feat, const int* idx3, const float* w3, float* out) {
  long t = (long)blockIdx.x * 256 + threadIdx.x;
  long total = (long)BB * NN * 768;
  if (t >= total) return;
  int c = (int)(t % 768);
  long bn = t / 768;
  int b = (int)(bn / NN);
  long o3 = bn * 3;
  float acc = 0.f;
  #pragma unroll
  for (int j = 0; j < 3; ++j)
    acc += w3[o3 + j] * feat[((long)b * SS + idx3[o3 + j]) * 768 + c];
  out[t] = acc;
}

__global__ void semean_kernel(const float* h, float* out) {
  int b = blockIdx.x;
  int c = blockIdx.y * 256 + threadIdx.x;    // 512
  if (c >= 512) return;
  float sm = 0.f;
  for (int n = 0; n < NN; ++n) sm += h[((long)b * NN + n) * 512 + c];
  out[b * 512 + c] = sm * (1.f / NN);
}

__global__ void gate_kernel(float* h, const float* se) {
  long t = (long)blockIdx.x * 256 + threadIdx.x;
  long total = (long)BB * NN * 512;
  if (t >= total) return;
  int c = (int)(t & 511);
  int b = (int)((t >> 9) >> 11);
  h[t] *= se[b * 512 + c];
}

__global__ void transpose_out_kernel(const float* h, float* out) {
  long t = (long)blockIdx.x * 256 + threadIdx.x;
  long total = (long)BB * 50 * NN;
  if (t >= total) return;
  int n = (int)(t % NN); long r = t / NN; int c = (int)(r % 50); int b = (int)(r / 50);
  out[t] = h[((long)b * NN + n) * 50 + c];
}

// ---------------------------------------------------------------------------
// Host orchestration
// ---------------------------------------------------------------------------
static GemmParams GP(const float* A, long lda, const float* Bw, long ldb,
                     float* C, long ldc, int M, int N, int K) {
  GemmParams p; memset(&p, 0, sizeof(p));
  p.A = A; p.lda = lda; p.Bw = Bw; p.ldb = ldb; p.C = C; p.ldc = ldc;
  p.M = M; p.N = N; p.K = K; p.zdiv = 1; p.alpha = 1.f;
  p.res_div = 1; p.amax_div = 1;
  return p;
}

struct BN4 { const float *g, *b, *m, *v; };
static void setBN(GemmParams& p, const BN4& bn) { p.bng = bn.g; p.bnb = bn.b; p.bnm = bn.m; p.bnv = bn.v; }

static void gemm(hipStream_t st, const GemmParams& p, int batches = 1) {
  dim3 grid((p.N + 63) / 64, (p.M + 63) / 64, batches);
  gemm_wmma_kernel<<<grid, dim3(128), 0, st>>>(p);
}

static inline dim3 g1d(long total) { return dim3((unsigned)((total + 255) / 256)); }

extern "C" void kernel_launch(void* const* d_in, const int* in_sizes, int n_in,
                              void* d_out, int out_size, void* d_ws, size_t ws_size,
                              hipStream_t stream) {
  (void)in_sizes; (void)n_in; (void)out_size; (void)ws_size;

  // ---- input mapping (setup_inputs dict insertion order) ----
  int ii = 0;
  auto nf = [&]() -> const float* { return (const float*)d_in[ii++]; };
  auto nbn = [&]() -> BN4 { BN4 r; r.g = nf(); r.b = nf(); r.m = nf(); r.v = nf(); return r; };

  const float* X = nf();              // x [16,3,2048]
  const float* L = nf();              // l [16,16]
  const float* conv1_w = nf(); BN4 bn1 = nbn();
  const float* conv2_w = nf(); BN4 bn2 = nbn();
  const float* lop_w1 = nf(); BN4 lop_bn1 = nbn();
  const float* lop_w2 = nf(); BN4 lop_bn2 = nbn();
  struct BlkP { const float *ln1g,*ln1b,*qkvw,*projw,*projb,*ln2g,*ln2b,*fc1w,*fc1b,*fc2w,*fc2b; } blk[8];
  for (int i = 0; i < 8; ++i) {
    blk[i].ln1g = nf(); blk[i].ln1b = nf();
    blk[i].qkvw = nf(); blk[i].projw = nf(); blk[i].projb = nf();
    blk[i].ln2g = nf(); blk[i].ln2b = nf();
    blk[i].fc1w = nf(); blk[i].fc1b = nf();
    blk[i].fc2w = nf(); blk[i].fc2b = nf();
  }
  const float* norm_g = nf(); const float* norm_b = nf();
  const float* red_w[3]; for (int i = 0; i < 3; ++i) red_w[i] = nf();
  BN4 red_bn[3];          for (int i = 0; i < 3; ++i) red_bn[i] = nbn();
  const float* label_w = nf(); BN4 label_bn = nbn();
  const float* fp_w1 = nf(); const float* fp_b1 = nf(); BN4 fp_bn1 = nbn();
  const float* fp_w2 = nf(); const float* fp_b2 = nf(); BN4 fp_bn2 = nbn();
  const float* h1_w = nf(); const float* h1_b = nf(); BN4 h_bn1 = nbn();
  const float* se_w1 = nf(); BN4 se_bn = nbn(); const float* se_w2 = nf();
  const float* h2_w = nf(); const float* h2_b = nf(); BN4 h_bn2 = nbn();
  const float* h3_w = nf(); const float* h3_b = nf();

  // ---- workspace layout ----
  size_t off = 0;
  auto alloc = [&](size_t elems, size_t esz) -> char* {
    off = (off + 255) & ~(size_t)255;
    char* p = (char*)d_ws + off;
    off += elems * esz;
    return p;
  };
  auto af = [&](size_t e) -> float* { return (float*)alloc(e, 4); };
  auto ai = [&](size_t e) -> int*   { return (int*)alloc(e, 4); };

  float* xyz     = af((size_t)BB * NN * 3);
  float* feats_a = af((size_t)BB * NN * 64);
  float* feats   = af((size_t)BB * NN * 64);
  int*   fps_idx = ai((size_t)BB * SS);
  float* new_xyz = af((size_t)BB * SS * 3);
  float* np_all  = af((size_t)BB * SS * 64);
  float* rvec    = af((size_t)BB * SS * 256);
  float* w1d     = af((size_t)256 * 64);
  int*   knn_idx = ai((size_t)BB * SS * SS);
  float* ga      = af(2L * SS * SS * 64);       // per-chunk gathered feats
  float* g1buf   = af(2L * SS * SS * 256);      // per-chunk Local_op hidden
  float* tok     = af((size_t)BB * SS * 256);
  float* xrun    = af((size_t)BB * SS * 256);
  float* save4   = af((size_t)BB * SS * 256);
  float* save8   = af((size_t)BB * SS * 256);
  float* lnb     = af((size_t)BB * SS * 256);
  float* qkvb    = af((size_t)BB * SS * 768);
  float* scores  = af((size_t)BB * HH * SS * SS);
  float* attno   = af((size_t)BB * SS * 256);
  float* mlph    = af((size_t)BB * SS * 512);
  float* feat768 = af((size_t)BB * SS * 768);
  float* fmx     = af((size_t)BB * 768);
  float* favg    = af((size_t)BB * 768);
  float* lab     = af((size_t)BB * 64);
  float* hb      = af((size_t)BB * 512);
  int*   idx3    = ai((size_t)BB * NN * 3);
  float* w3      = af((size_t)BB * NN * 3);
  float* fp1     = af((size_t)BB * NN * 1024);
  float* fp2     = af((size_t)BB * NN * 1024);
  float* h1o     = af((size_t)BB * NN * 512);
  float* sem     = af((size_t)BB * 512);
  float* se1     = af((size_t)BB * 64);
  float* se2     = af((size_t)BB * 512);
  float* h2o     = af((size_t)BB * NN * 256);
  float* interp  = ga;       // alias: ga+g1buf region free after Local_op
  float* h3o     = scores;   // alias: scores free after transformer

  const long MN = (long)BB * NN;      // 32768 rows
  const long MS = (long)BB * SS;      // 4096 rows

  // ---- stem: transpose + conv1 + conv2 (channels-last everywhere) ----
  transpose_x_kernel<<<g1d((long)BB * 3 * NN), 256, 0, stream>>>(X, xyz);
  { GemmParams p = GP(xyz, 3, conv1_w, 3, feats_a, 64, (int)MN, 64, 3);
    setBN(p, bn1); p.act = 2; gemm(stream, p); }
  { GemmParams p = GP(feats_a, 64, conv2_w, 64, feats, 64, (int)MN, 64, 64);
    setBN(p, bn2); p.act = 2; gemm(stream, p); }

  // ---- FPS + kNN grouping ----
  fps_kernel<<<BB, 256, 0, stream>>>(xyz, fps_idx);
  gather_fps_kernel<<<g1d(BB * SS), 256, 0, stream>>>(xyz, feats, fps_idx, new_xyz, np_all);
  knn_kernel<<<dim3(SS, BB), 256, 0, stream>>>(xyz, new_xyz, knn_idx);

  // ---- Local_op: g1 = relu(bn1(gather@W1a^T + np@(W1b-W1a)^T)); tok = max_k relu(bn2(g1@W2^T)) ----
  w1diff_kernel<<<g1d(256 * 64), 256, 0, stream>>>(lop_w1, w1d);
  { GemmParams p = GP(np_all, 64, w1d, 64, rvec, 256, (int)MS, 256, 64); gemm(stream, p); }
  zero_kernel<<<g1d(MS * 256), 256, 0, stream>>>(tok, MS * 256);
  for (int c = 0; c < 8; ++c) {
    int b0 = 2 * c;
    gather_knn_kernel<<<g1d(2L * SS * SS * 64), 256, 0, stream>>>(feats, knn_idx, ga, b0);
    { GemmParams p = GP(ga, 64, lop_w1, 128, g1buf, 256, 2 * SS * SS, 256, 64);
      setBN(p, lop_bn1); p.act = 1;
      p.res = rvec + (long)b0 * SS * 256; p.ldres = 256; p.resmode = 2; p.res_div = 256;
      gemm(stream, p); }
    { GemmParams p = GP(g1buf, 256, lop_w2, 256, tok + (long)b0 * SS * 256, 256,
                        2 * SS * SS, 256, 256);
      setBN(p, lop_bn2); p.act = 1; p.amax = 1; p.amax_div = 256;
      gemm(stream, p); }
  }

  // ---- transformer blocks ----
  hipMemcpyAsync(xrun, tok, MS * 256 * 4, hipMemcpyDeviceToDevice, stream);
  for (int i = 0; i < 8; ++i) {
    layernorm_kernel<<<(int)MS, 256, 0, stream>>>(xrun, lnb, blk[i].ln1g, blk[i].ln1b);
    { GemmParams p = GP(lnb, 256, blk[i].qkvw, 256, qkvb, 768, (int)MS, 768, 256);
      gemm(stream, p); }
    { GemmParams p = GP(qkvb, 768, qkvb + 256, 768, scores, 256, SS, SS, DHH);
      p.zdiv = HH; p.alpha = 0.125f;
      p.sAo = (long)SS * 768; p.sAi = DHH;
      p.sBo = (long)SS * 768; p.sBi = DHH;
      p.sCo = (long)HH * SS * SS; p.sCi = (long)SS * SS;
      gemm(stream, p, BB * HH); }
    softmax256_kernel<<<BB * HH * SS, 256, 0, stream>>>(scores);
    { GemmParams p = GP(scores, 256, qkvb + 512, 768, attno, 256, SS, DHH, SS);
      p.b_kn = 1; p.zdiv = HH;
      p.sAo = (long)HH * SS * SS; p.sAi = (long)SS * SS;
      p.sBo = (long)SS * 768; p.sBi = DHH;
      p.sCo = (long)SS * 256; p.sCi = DHH;
      gemm(stream, p, BB * HH); }
    { GemmParams p = GP(attno, 256, blk[i].projw, 256, xrun, 256, (int)MS, 256, 256);
      p.bias = blk[i].projb; p.res = xrun; p.ldres = 256; p.resmode = 1;
      gemm(stream, p); }
    layernorm_kernel<<<(int)MS, 256, 0, stream>>>(xrun, lnb, blk[i].ln2g, blk[i].ln2b);
    { GemmParams p = GP(lnb, 256, blk[i].fc1w, 256, mlph, 512, (int)MS, 512, 256);
      p.bias = blk[i].fc1b; p.act = 3; gemm(stream, p); }
    { GemmParams p = GP(mlph, 512, blk[i].fc2w, 512, xrun, 256, (int)MS, 256, 512);
      p.bias = blk[i].fc2b; p.res = xrun; p.ldres = 256; p.resmode = 1;
      gemm(stream, p); }
    if (i == 3) hipMemcpyAsync(save4, xrun, MS * 256 * 4, hipMemcpyDeviceToDevice, stream);
    if (i == 7) hipMemcpyAsync(save8, xrun, MS * 256 * 4, hipMemcpyDeviceToDevice, stream);
  }

  // ---- reductions: LN(norm) + red conv, concat into feat768 column bands ----
  const float* srcs[3] = { tok, save4, save8 };
  for (int i = 0; i < 3; ++i) {
    layernorm_kernel<<<(int)MS, 256, 0, stream>>>(srcs[i], lnb, norm_g, norm_b);
    GemmParams p = GP(lnb, 256, red_w[i], 256, feat768 + 256 * i, 768, (int)MS, 256, 256);
    setBN(p, red_bn[i]); p.act = 2; gemm(stream, p);
  }

  // ---- global pooling + label branch ----
  pool_kernel<<<dim3(BB, 3), 256, 0, stream>>>(feat768, fmx, favg);
  { GemmParams p = GP(L, 16, label_w, 16, lab, 64, BB, 64, 16);
    setBN(p, label_bn); p.act = 2; gemm(stream, p); }

  // ---- 3-NN inverse-distance interpolation ----
  three_nn_kernel<<<dim3(BB, NN / 256), 256, 0, stream>>>(xyz, new_xyz, idx3, w3);
  interp_kernel<<<g1d(MN * 768), 256, 0, stream>>>(feat768, idx3, w3, interp);

  // ---- feature propagation convs (split-K over concat [x_local | interp]) ----
  { GemmParams p = GP(feats, 64, fp_w1, 832, fp1, 1024, (int)MN, 1024, 64);
    gemm(stream, p); }
  { GemmParams p = GP(interp, 768, fp_w1 + 64, 832, fp1, 1024, (int)MN, 1024, 768);
    p.load_c = 1; p.bias = fp_b1; setBN(p, fp_bn1); p.act = 1; gemm(stream, p); }
  { GemmParams p = GP(fp1, 1024, fp_w2, 1024, fp2, 1024, (int)MN, 1024, 1024);
    p.bias = fp_b2; setBN(p, fp_bn2); p.act = 1; gemm(stream, p); }

  // ---- head: per-batch global term hb = xg @ h1_w[:,1024:]^T, then h1 over fp2 ----
  { GemmParams p = GP(fmx, 768, h1_w + 1024, 2624, hb, 512, BB, 512, 768); gemm(stream, p); }
  { GemmParams p = GP(favg, 768, h1_w + 1792, 2624, hb, 512, BB, 512, 768);
    p.load_c = 1; gemm(stream, p); }
  { GemmParams p = GP(lab, 64, h1_w + 2560, 2624, hb, 512, BB, 512, 64);
    p.load_c = 1; gemm(stream, p); }
  { GemmParams p = GP(fp2, 1024, h1_w, 2624, h1o, 512, (int)MN, 512, 1024);
    p.bias = h1_b; p.res = hb; p.ldres = 512; p.resmode = 2; p.res_div = NN;
    setBN(p, h_bn1); p.act = 2; gemm(stream, p); }

  // ---- SE gating ----
  semean_kernel<<<dim3(BB, 2), 256, 0, stream>>>(h1o, sem);
  { GemmParams p = GP(sem, 512, se_w1, 512, se1, 64, BB, 64, 512);
    setBN(p, se_bn); p.act = 2; gemm(stream, p); }
  { GemmParams p = GP(se1, 64, se_w2, 64, se2, 512, BB, 512, 64);
    p.act = 4; gemm(stream, p); }
  gate_kernel<<<g1d(MN * 512), 256, 0, stream>>>(h1o, se2);

  // ---- final classifier ----
  { GemmParams p = GP(h1o, 512, h2_w, 512, h2o, 256, (int)MN, 256, 512);
    p.bias = h2_b; setBN(p, h_bn2); p.act = 2; gemm(stream, p); }
  { GemmParams p = GP(h2o, 256, h3_w, 256, h3o, 50, (int)MN, 50, 256);
    p.bias = h3_b; gemm(stream, p); }
  transpose_out_kernel<<<g1d((long)BB * 50 * NN), 256, 0, stream>>>(h3o, (float*)d_out);
}
